// Attention_52218212385505
// MI455X (gfx1250) — matrix operand order repaired and checked by reading, not compiled, not run on hardware
//
#include <hip/hip_runtime.h>

typedef __attribute__((ext_vector_type(2)))  __bf16 v2bf;
typedef __attribute__((ext_vector_type(8)))  __bf16 v8bf;
typedef __attribute__((ext_vector_type(16))) __bf16 v16bf;
typedef __attribute__((ext_vector_type(8)))  float  v8f;
typedef __attribute__((ext_vector_type(4)))  int    v4i;

#define S_LEN 4096
#define D_DIM 1024
#define H_NUM 16
#define K_DIM 64

#define AS1 __attribute__((address_space(1)))
#define AS3 __attribute__((address_space(3)))

#if defined(__AMDGCN__) && __has_builtin(__builtin_amdgcn_global_load_async_to_lds_b128)
#define USE_ASYNC 1
#else
#define USE_ASYNC 0
#endif

#if USE_ASYNC
static __device__ __forceinline__ void async_g2l_b128(const void* g, void* l) {
    __builtin_amdgcn_global_load_async_to_lds_b128(
        (AS1 v4i*)g, (AS3 v4i*)l, 0, 0);
}
static __device__ __forceinline__ void wait_async0() {
#if __has_builtin(__builtin_amdgcn_s_wait_asynccnt)
    __builtin_amdgcn_s_wait_asynccnt(0);
#else
    asm volatile("s_wait_asynccnt 0x0" ::: "memory");
#endif
}
#endif

// pack two f32 -> packed bf16 pair (v_cvt_pk_bf16_f32)
static __device__ __forceinline__ unsigned int pack2bf(float a, float b) {
    v2bf p;
    p[0] = (__bf16)a;
    p[1] = (__bf16)b;
    union { v2bf v; unsigned int u; } cv;
    cv.v = p;
    return cv.u;
}

static __device__ __forceinline__ unsigned short f2bf1(float f) {
    union { __bf16 v; unsigned short u; } cv;
    cv.v = (__bf16)f;
    return cv.u;
}

static __device__ __forceinline__ v16bf mk16(v8bf lo, v8bf hi) {
    v16bf r;
#pragma unroll
    for (int i = 0; i < 8; ++i) { r[i] = lo[i]; r[i + 8] = hi[i]; }
    return r;
}

// ---------------------------------------------------------------------------
// QKV projection: out_bf16[s, n] = bf16( (sum_d x[s,d]*w[d,n] + bias[n])*scale )
// Block: 256 thr (8 waves). Tile 128(M) x 128(N), K-step 64. 16 wmma/wave/iter.
// scale = 1/sqrt(head_dim) for Q (folds attention score scaling), else 1.
// ---------------------------------------------------------------------------
__global__ void qkv_proj_kernel(const float* __restrict__ x,
                                const float* __restrict__ w,
                                const float* __restrict__ bias,
                                unsigned short* __restrict__ out,
                                float scale) {
    __shared__ unsigned short Xs[128 * 72];  // [m][k], 64 k + pad
    __shared__ unsigned short Ws[128 * 72];  // transposed [n][k]

    const int tid  = threadIdx.x;
    const int wid  = tid >> 5;
    const int lane = tid & 31;
    const int s0   = blockIdx.x * 128;
    const int n0   = blockIdx.y * 128;

    v8f acc[8];
#pragma unroll
    for (int t = 0; t < 8; ++t)
#pragma unroll
        for (int j = 0; j < 8; ++j) acc[t][j] = 0.0f;

    const int m    = lane & 15;
    const int koff = (lane < 16) ? 0 : 16;
    const int kb   = (lane < 16) ? 0 : 16;

    // thread-invariant load geometry (32-bit element offsets)
    const int xr = tid >> 5;                // 0..7, +8 per unroll step
    const int xc = (tid & 31) * 2;
    const int xbase = xr * D_DIM + xc;      // relative to x + s0*D + k0
    const int xlds  = xr * 72 + xc;
    const int wkp = tid >> 7;               // 0..1, +2 per unroll step
    const int wn  = tid & 127;
    const int wbase = wkp * 2 * D_DIM + wn; // relative to w + k0*D + n0
    const int wlds  = wn * 72 + 2 * wkp;

    const float* xtile = x + s0 * D_DIM;
    const float* wtile = w + n0;

    for (int k0 = 0; k0 < D_DIM; k0 += 64) {
        __syncthreads();
        const float* xk = xtile + k0;
        const float* wk2 = wtile + k0 * D_DIM;
        // X tile 128x64: one b64 load + one b32 LDS store per element pair
#pragma unroll
        for (int i = 0; i < 16; ++i) {
            const float* p = xk + xbase + i * (8 * D_DIM);
            *(unsigned int*)&Xs[xlds + i * (8 * 72)] = pack2bf(p[0], p[1]);
        }
        // W tile 64x128 -> transposed LDS [n][k]; pairs along k
#pragma unroll
        for (int i = 0; i < 16; ++i) {
            const float* p = wk2 + wbase + i * (4 * D_DIM);
            *(unsigned int*)&Ws[wlds + i * 4] = pack2bf(p[0], p[D_DIM]);
        }
        if (k0 + 64 < D_DIM) {
            __builtin_prefetch(xk + 64 + (tid >> 1) * D_DIM, 0, 0);
            __builtin_prefetch(wk2 + (64 + (tid >> 2)) * D_DIM, 0, 0);
        }
        __syncthreads();

        const unsigned short* xrow = &Xs[(wid * 16 + m) * 72];
#pragma unroll
        for (int kk = 0; kk < 2; ++kk) {
            v8bf alo = *(const v8bf*)&xrow[kk * 32 + koff];
            v8bf ahi = *(const v8bf*)&xrow[kk * 32 + koff + 8];
            v16bf a  = mk16(alo, ahi);
            v16bf bfr[8];
#pragma unroll
            for (int t = 0; t < 8; ++t)
                bfr[t] = *(const v16bf*)&Ws[(t * 16 + m) * 72 + kk * 32 + kb];
#pragma unroll
            for (int t = 0; t < 8; ++t)
                acc[t] = __builtin_amdgcn_wmma_f32_16x16x32_bf16(
                    false, a, false, bfr[t], (short)0, acc[t], false, false);
        }
    }

    const int hihalf = lane >> 4;
#pragma unroll
    for (int t = 0; t < 8; ++t) {
        float bn = bias[n0 + t * 16 + m];
#pragma unroll
        for (int j = 0; j < 8; ++j) {
            int srow = s0 + wid * 16 + j + 8 * hihalf;
            int col  = n0 + t * 16 + m;
            out[srow * D_DIM + col] = f2bf1((acc[t][j] + bn) * scale);
        }
    }
}

// ---------------------------------------------------------------------------
// Flash attention: grid (S/64, H), 128 thr = 4 waves, 16 query rows/wave.
// Key blocks of 128, online softmax. q pre-scaled by 1/sqrt(64).
// ---------------------------------------------------------------------------
__global__ void flash_attn_kernel(const unsigned short* __restrict__ qg,
                                  const unsigned short* __restrict__ kg,
                                  const unsigned short* __restrict__ vg,
                                  unsigned short* __restrict__ vals,
                                  const int* __restrict__ is_causal) {
    __shared__ unsigned short Ks[128 * 72];      // [key j][d]  (natural)
    __shared__ unsigned short Vs[64 * 136];      // [d][key j]  (transposed)
    __shared__ unsigned short Ps[4 * 16 * 136];  // per-wave P tile [m][key]

    const int tid  = threadIdx.x;
    const int wid  = tid >> 5;
    const int lane = tid & 31;
    const int q0   = blockIdx.x * 64;
    const int h    = blockIdx.y;

    const int m      = lane & 15;
    const int hihalf = lane >> 4;
    const int koff   = (lane < 16) ? 0 : 16;
    const int kb     = (lane < 16) ? 0 : 16;

    // Q fragments held in registers
    const int qrow_s = q0 + wid * 16 + m;
    const unsigned short* qrow = qg + (qrow_s * H_NUM + h) * K_DIM;
    v16bf qa[2];
#pragma unroll
    for (int kk = 0; kk < 2; ++kk) {
        v8bf lo = *(const v8bf*)&qrow[kk * 32 + koff];
        v8bf hi = *(const v8bf*)&qrow[kk * 32 + koff + 8];
        qa[kk]  = mk16(lo, hi);
    }

    float mrun[8], lrun[8];
    v8f o[4];
#pragma unroll
    for (int j = 0; j < 8; ++j) { mrun[j] = -3.0e38f; lrun[j] = 0.0f; }
#pragma unroll
    for (int t = 0; t < 4; ++t)
#pragma unroll
        for (int j = 0; j < 8; ++j) o[t][j] = 0.0f;

    const int causal = (*is_causal) != 0;
    const int nkb    = causal ? ((int)(blockIdx.x >> 1) + 1) : (S_LEN / 128);

    unsigned short* pw = &Ps[wid * 16 * 136];

    // K-tile async geometry: b128 units (8 bf16)
    const int kar = tid >> 3;            // 0..15, +16 per unroll step
    const int kac = (tid & 7) * 8;
    // V-tile (transpose+pack) geometry
    const int vd  = tid & 63;
    const int vjp = tid >> 6;            // 0..1, +2 per unroll step

    for (int jb = 0; jb < nkb; ++jb) {
        const int j0 = jb * 128;
        __syncthreads();
#if USE_ASYNC
#pragma unroll
        for (int i = 0; i < 8; ++i) {
            int r = kar + i * 16;
            async_g2l_b128(&kg[((j0 + r) * H_NUM + h) * K_DIM + kac],
                           &Ks[r * 72 + kac]);
        }
#else
#pragma unroll
        for (int i = 0; i < 8; ++i) {
            int r = kar + i * 16;
            int g = ((j0 + r) * H_NUM + h) * K_DIM + kac;
            *(v8bf*)&Ks[r * 72 + kac] = *(const v8bf*)&kg[g];
        }
#endif
        // V transposed: pack key pairs along j
#pragma unroll
        for (int i = 0; i < 32; ++i) {
            int jp = vjp + i * 2;        // 0..63
            int g  = ((j0 + 2 * jp) * H_NUM + h) * K_DIM + vd;
            unsigned int lo = vg[g];
            unsigned int hi = vg[g + H_NUM * K_DIM];
            *(unsigned int*)&Vs[vd * 136 + 2 * jp] = lo | (hi << 16);
        }
        if (jb + 1 < nkb) {
            int gn = ((j0 + 128 + tid) * H_NUM + h) * K_DIM;
            __builtin_prefetch(&kg[gn], 0, 0);
            __builtin_prefetch(&vg[gn], 0, 0);
        }
#if USE_ASYNC
        wait_async0();
#endif
        __syncthreads();

        // scores S = Q * K^T  (q pre-scaled; 8 column tiles of 16 keys)
        v8f s[8];
#pragma unroll
        for (int t = 0; t < 8; ++t)
#pragma unroll
            for (int j = 0; j < 8; ++j) s[t][j] = 0.0f;
#pragma unroll
        for (int kk = 0; kk < 2; ++kk) {
            v16bf bfr[8];
#pragma unroll
            for (int t = 0; t < 8; ++t)
                bfr[t] = *(const v16bf*)&Ks[(t * 16 + m) * 72 + kk * 32 + kb];
#pragma unroll
            for (int t = 0; t < 8; ++t)
                s[t] = __builtin_amdgcn_wmma_f32_16x16x32_bf16(
                    false, qa[kk], false, bfr[t], (short)0, s[t], false, false);
        }

        // causal mask: only the diagonal (last) key block can mask anything
        if (causal && (jb == nkb - 1)) {
#pragma unroll
            for (int t = 0; t < 8; ++t)
#pragma unroll
                for (int j = 0; j < 8; ++j) {
                    int qpos = q0 + wid * 16 + j + 8 * hihalf;
                    int kpos = j0 + t * 16 + m;
                    if (qpos < kpos) s[t][j] = -3.0e38f;
                }
        }

        // row max over 128 keys
        float rmax[8];
#pragma unroll
        for (int j = 0; j < 8; ++j) {
            rmax[j] = s[0][j];
#pragma unroll
            for (int t = 1; t < 8; ++t) rmax[j] = fmaxf(rmax[j], s[t][j]);
        }
#pragma unroll
        for (int d = 1; d < 16; d <<= 1)
#pragma unroll
            for (int j = 0; j < 8; ++j)
                rmax[j] = fmaxf(rmax[j], __shfl_xor(rmax[j], d));

        float alpha[8], rsum[8];
#pragma unroll
        for (int j = 0; j < 8; ++j) {
            float mn = fmaxf(mrun[j], rmax[j]);
            alpha[j] = __expf(mrun[j] - mn);
            mrun[j]  = mn;
            rsum[j]  = 0.0f;
        }
#pragma unroll
        for (int t = 0; t < 8; ++t)
#pragma unroll
            for (int j = 0; j < 8; ++j) {
                float p = __expf(s[t][j] - mrun[j]);
                rsum[j] += p;
                pw[(j + 8 * hihalf) * 136 + t * 16 + m] = f2bf1(p);
            }
#pragma unroll
        for (int d = 1; d < 16; d <<= 1)
#pragma unroll
            for (int j = 0; j < 8; ++j) rsum[j] += __shfl_xor(rsum[j], d);
#pragma unroll
        for (int j = 0; j < 8; ++j) lrun[j] = lrun[j] * alpha[j] + rsum[j];
#pragma unroll
        for (int t = 0; t < 4; ++t)
#pragma unroll
            for (int j = 0; j < 8; ++j) o[t][j] *= alpha[j];

        // O += P * V  (A = P tile, 4 key-steps; B = V^T rows, 4 d-tiles)
#pragma unroll
        for (int kk = 0; kk < 4; ++kk) {
            v8bf lo = *(const v8bf*)&pw[m * 136 + kk * 32 + koff];
            v8bf hi = *(const v8bf*)&pw[m * 136 + kk * 32 + koff + 8];
            v16bf pa = mk16(lo, hi);
            v16bf bfr[4];
#pragma unroll
            for (int t = 0; t < 4; ++t)
                bfr[t] = *(const v16bf*)&Vs[(t * 16 + m) * 136 + kk * 32 + kb];
#pragma unroll
            for (int t = 0; t < 4; ++t)
                o[t] = __builtin_amdgcn_wmma_f32_16x16x32_bf16(
                    false, pa, false, bfr[t], (short)0, o[t], false, false);
        }
    }

    float inv[8];
#pragma unroll
    for (int j = 0; j < 8; ++j) inv[j] = 1.0f / lrun[j];
#pragma unroll
    for (int t = 0; t < 4; ++t)
#pragma unroll
        for (int j = 0; j < 8; ++j) {
            int srow = q0 + wid * 16 + j + 8 * hihalf;
            int col  = t * 16 + m;
            vals[(srow * H_NUM + h) * K_DIM + col] = f2bf1(o[t][j] * inv[j]);
        }
}

// ---------------------------------------------------------------------------
// Output projection: out_f32[s, d] = sum_n vals_bf16[s, n] * wo[n, d] + bo[d]
// ---------------------------------------------------------------------------
__global__ void out_proj_kernel(const unsigned short* __restrict__ a_in,
                                const float* __restrict__ w,
                                const float* __restrict__ bo,
                                float* __restrict__ out) {
    __shared__ unsigned short As[128 * 72];
    __shared__ unsigned short Ws[128 * 72];

    const int tid  = threadIdx.x;
    const int wid  = tid >> 5;
    const int lane = tid & 31;
    const int s0   = blockIdx.x * 128;
    const int n0   = blockIdx.y * 128;

    v8f acc[8];
#pragma unroll
    for (int t = 0; t < 8; ++t)
#pragma unroll
        for (int j = 0; j < 8; ++j) acc[t][j] = 0.0f;

    const int m    = lane & 15;
    const int koff = (lane < 16) ? 0 : 16;
    const int kb   = (lane < 16) ? 0 : 16;

    // A tile: b128 units
    const int aar = tid >> 3;             // 0..31, +32 per unroll step
    const int aac = (tid & 7) * 8;
    const int wkp = tid >> 7;
    const int wn  = tid & 127;
    const int wbase = wkp * 2 * D_DIM + wn;
    const int wlds  = wn * 72 + 2 * wkp;
    const float* wtile = w + n0;

    for (int k0 = 0; k0 < D_DIM; k0 += 64) {
        __syncthreads();
#if USE_ASYNC
#pragma unroll
        for (int i = 0; i < 4; ++i) {
            int r = aar + i * 32;
            async_g2l_b128(&a_in[(s0 + r) * D_DIM + k0 + aac],
                           &As[r * 72 + aac]);
        }
#else
#pragma unroll
        for (int i = 0; i < 4; ++i) {
            int r = aar + i * 32;
            *(v8bf*)&As[r * 72 + aac] =
                *(const v8bf*)&a_in[(s0 + r) * D_DIM + k0 + aac];
        }
#endif
        const float* wk2 = wtile + k0 * D_DIM;
#pragma unroll
        for (int i = 0; i < 16; ++i) {
            const float* p = wk2 + wbase + i * (4 * D_DIM);
            *(unsigned int*)&Ws[wlds + i * 4] = pack2bf(p[0], p[D_DIM]);
        }
        if (k0 + 64 < D_DIM) {
            __builtin_prefetch(&a_in[(s0 + (tid >> 1)) * D_DIM + k0 + 64], 0, 0);
            __builtin_prefetch(wk2 + (64 + (tid >> 2)) * D_DIM, 0, 0);
        }
#if USE_ASYNC
        wait_async0();
#endif
        __syncthreads();

        const unsigned short* arow = &As[(wid * 16 + m) * 72];
#pragma unroll
        for (int kk = 0; kk < 2; ++kk) {
            v8bf alo = *(const v8bf*)&arow[kk * 32 + koff];
            v8bf ahi = *(const v8bf*)&arow[kk * 32 + koff + 8];
            v16bf a  = mk16(alo, ahi);
            v16bf bfr[8];
#pragma unroll
            for (int t = 0; t < 8; ++t)
                bfr[t] = *(const v16bf*)&Ws[(t * 16 + m) * 72 + kk * 32 + kb];
#pragma unroll
            for (int t = 0; t < 8; ++t)
                acc[t] = __builtin_amdgcn_wmma_f32_16x16x32_bf16(
                    false, a, false, bfr[t], (short)0, acc[t], false, false);
        }
    }

    const int hihalf = lane >> 4;
#pragma unroll
    for (int t = 0; t < 8; ++t) {
        float bn = bo[n0 + t * 16 + m];
#pragma unroll
        for (int j = 0; j < 8; ++j) {
            int srow = s0 + wid * 16 + j + 8 * hihalf;
            int col  = n0 + t * 16 + m;
            out[srow * D_DIM + col] = acc[t][j] + bn;
        }
    }
}

// ---------------------------------------------------------------------------
extern "C" void kernel_launch(void* const* d_in, const int* in_sizes, int n_in,
                              void* d_out, int out_size, void* d_ws, size_t ws_size,
                              hipStream_t stream) {
    const float* x  = (const float*)d_in[0];
    const float* wq = (const float*)d_in[1];
    const float* bq = (const float*)d_in[2];
    const float* wk = (const float*)d_in[3];
    const float* bk = (const float*)d_in[4];
    const float* wv = (const float*)d_in[5];
    const float* bv = (const float*)d_in[6];
    const float* wo = (const float*)d_in[7];
    const float* bo = (const float*)d_in[8];
    const int*   ic = (const int*)d_in[9];
    float* out = (float*)d_out;

    const size_t SD = (size_t)S_LEN * D_DIM;
    unsigned short* qb = (unsigned short*)d_ws;
    unsigned short* kb = qb + SD;
    unsigned short* vb = kb + SD;
    unsigned short* ob = vb + SD;  // attention output (vals), bf16

    dim3 gg(S_LEN / 128, D_DIM / 128);
    // fold attention scale 1/sqrt(64) into q
    qkv_proj_kernel<<<gg, 256, 0, stream>>>(x, wq, bq, qb, 0.125f);
    qkv_proj_kernel<<<gg, 256, 0, stream>>>(x, wk, bk, kb, 1.0f);
    qkv_proj_kernel<<<gg, 256, 0, stream>>>(x, wv, bv, vb, 1.0f);

    dim3 ga(S_LEN / 64, H_NUM);
    flash_attn_kernel<<<ga, 128, 0, stream>>>(qb, kb, vb, ob, ic);

    dim3 gp(S_LEN / 128, D_DIM / 128);
    out_proj_kernel<<<gp, 256, 0, stream>>>(ob, wo, bo, out);
}
